// ProbAttn1_38723425141443
// MI455X (gfx1250) — compile-verified, matrix operand order checked by reading, not run on hardware
//
#include <hip/hip_runtime.h>
#include <math.h>

typedef __attribute__((ext_vector_type(16))) _Float16 v16h;
typedef __attribute__((ext_vector_type(8)))  float    v8f;

#define B_   4
#define L_   2048
#define H_   8
#define D_   64
#define SK   40   // sample_k
#define U_   40   // top-u queries kept
#define UPAD 48   // padded to 3 tiles of 16
#define MT   3    // number of 16-row m-tiles

// K-dim index held by fragment element e for the 16-bit A/B WMMA layout
// (CDNA5 ISA, 16-bit A-Matrix 16x32 table; B mirrors with lane->N).
__device__ __forceinline__ int frag_k(int e, int hi) {
  int v = e >> 1;
  int base = (v < 4) ? (v << 1) : (16 + ((v - 4) << 1));
  return base + (hi ? 8 : 0) + (e & 1);
}

// ---------------- Kernel 1: sparsity measure M = max - mean --------------
// One wave (32 lanes) per query; each lane owns 2 of the 64 channels.
__global__ void pa_measure_kernel(const float* __restrict__ q,
                                  const float* __restrict__ k,
                                  const int*   __restrict__ idx,
                                  float*       __restrict__ M) {
  const int lane = threadIdx.x & 31;
  const int wave = (blockIdx.x * blockDim.x + threadIdx.x) >> 5;
  if (wave >= B_ * H_ * L_) return;
  const int qi = wave % L_;
  const int bh = wave / L_;
  const int b = bh / H_, h = bh % H_;
  const float2 qv =
      *(const float2*)(q + (((size_t)(b * L_ + qi) * H_ + h) * D_) + lane * 2);
  float mx = -INFINITY, sm = 0.f;
  for (int s = 0; s < SK; ++s) {
    const int kr = idx[qi * SK + s];
    const float2 kv =
        *(const float2*)(k + (((size_t)(b * L_ + kr) * H_ + h) * D_) + lane * 2);
    float d = qv.x * kv.x + qv.y * kv.y;
#pragma unroll
    for (int m = 16; m >= 1; m >>= 1) d += __shfl_xor(d, m, 32);
    mx = fmaxf(mx, d);
    sm += d;
  }
  if (lane == 0) M[(size_t)bh * L_ + qi] = mx - sm * (1.0f / (float)L_);
}

// ---------------- Kernel 2: top-40 indices per (b,h) ---------------------
__global__ void pa_topk_kernel(const float* __restrict__ M,
                               int*         __restrict__ top) {
  __shared__ float vals[L_];
  __shared__ float rv[256];
  __shared__ int   ri[256];
  const int bh = blockIdx.x;
  const int t  = threadIdx.x;
  for (int i = t; i < L_; i += 256) vals[i] = M[(size_t)bh * L_ + i];
  __syncthreads();
  for (int it = 0; it < U_; ++it) {
    float bv = -INFINITY;
    int   bi = 0x7fffffff;
    for (int i = t; i < L_; i += 256) {
      float v = vals[i];
      if (v > bv || (v == bv && i < bi)) { bv = v; bi = i; }
    }
    rv[t] = bv; ri[t] = bi;
    __syncthreads();
    for (int s = 128; s > 0; s >>= 1) {
      if (t < s) {
        float ov = rv[t + s]; int oi = ri[t + s];
        if (ov > rv[t] || (ov == rv[t] && oi < ri[t])) { rv[t] = ov; ri[t] = oi; }
      }
      __syncthreads();
    }
    if (t == 0) { top[bh * U_ + it] = ri[0]; vals[ri[0]] = -INFINITY; }
    __syncthreads();
  }
}

// ---------------- Kernel 3: cumsum(V) -> ws laid out (B,H,L,D) ----------
__global__ void pa_cumsum_kernel(const float* __restrict__ val,
                                 float*       __restrict__ vc) {
  const int g = blockIdx.x * blockDim.x + threadIdx.x;
  if (g >= B_ * H_ * D_) return;
  const int b  = g / (H_ * D_);
  const int hd = g % (H_ * D_);
  const int h = hd / D_, d = hd % D_;
  float acc = 0.f;
  for (int l = 0; l < L_; ++l) {
    acc += val[((size_t)(b * L_ + l) * (H_ * D_)) + hd];
    vc[(((size_t)(b * H_ + h) * L_) + l) * D_ + d] = acc;
  }
}

// ---------------- Kernel 4: WMMA flash attention of top queries ----------
// One block per (b,h): 4 waves; waves 0..2 each own a 16-row query tile.
__global__ void __launch_bounds__(128)
pa_attn_kernel(const float* __restrict__ q,
               const float* __restrict__ k,
               const float* __restrict__ vc,
               const int*   __restrict__ top,
               float*       __restrict__ out) {
  __shared__ _Float16 qs[UPAD][D_];
  __shared__ _Float16 ks[32][D_];
  __shared__ _Float16 vs[32][D_];
  __shared__ _Float16 pb[4][16][32];

  const int bh = blockIdx.x;
  const int b = bh / H_, h = bh % H_;
  const int t    = threadIdx.x;
  const int wave = t >> 5;
  const int lane = t & 31;
  const int hi   = (lane >> 4) & 1;
  const int ln   = lane & 15;

  // Stage Q_reduce tile (48 rows, rows >= 40 zero-padded).
  for (int i = t; i < UPAD * D_; i += 128) {
    const int r = i / D_, d = i % D_;
    float v = 0.f;
    if (r < U_) {
      const int qi = top[bh * U_ + r];
      v = q[(((size_t)(b * L_ + qi) * H_ + h) * D_) + d];
    }
    qs[r][d] = (_Float16)v;
  }
  __syncthreads();

  // Q A-fragments (16x32 f16 each, two d-chunks), constant across key blocks.
  v16h aq0 = {}, aq1 = {};
  if (wave < MT) {
#pragma unroll
    for (int e = 0; e < 16; ++e) {
      const int kk = frag_k(e, hi);
      aq0[e] = qs[wave * 16 + ln][kk];
      aq1[e] = qs[wave * 16 + ln][kk + 32];
    }
  }

  float mrow[8], lsum[8];
  v8f oo[4] = {};
#pragma unroll
  for (int r = 0; r < 8; ++r) { mrow[r] = -INFINITY; lsum[r] = 0.f; }
  const float scale = 0.125f; // 1/sqrt(64)

  for (int kb = 0; kb < L_ / 32; ++kb) {
    // Cooperative stage: 32 keys x 64 ch of K and cumsum-V, f32 -> f16.
    for (int i = t; i < 32 * D_; i += 128) {
      const int r = i / D_, d = i % D_;
      const int l = kb * 32 + r;
      ks[r][d] = (_Float16)k[(((size_t)(b * L_ + l) * H_ + h) * D_) + d];
      vs[r][d] = (_Float16)vc[((size_t)bh * L_ + l) * D_ + d];
    }
    __syncthreads();

    if (wave < MT) {
      // ---- S = Q * K^T for this 32-key block: two 16x16 f32 tiles ----
      v8f s0 = {}, s1 = {};
      {
        v16h b0, b1;
#pragma unroll
        for (int e = 0; e < 16; ++e) {
          const int kk = frag_k(e, hi);       // kk = d index (matmul K-dim)
          b0[e] = ks[ln][kk];                 // keys 0..15  (N = lane)
          b1[e] = ks[ln][kk + 32];
        }
        s0 = __builtin_amdgcn_wmma_f32_16x16x32_f16(false, aq0, false, b0,
                                                    (short)0, s0, false, false);
        s0 = __builtin_amdgcn_wmma_f32_16x16x32_f16(false, aq1, false, b1,
                                                    (short)0, s0, false, false);
#pragma unroll
        for (int e = 0; e < 16; ++e) {
          const int kk = frag_k(e, hi);
          b0[e] = ks[16 + ln][kk];            // keys 16..31
          b1[e] = ks[16 + ln][kk + 32];
        }
        s1 = __builtin_amdgcn_wmma_f32_16x16x32_f16(false, aq0, false, b0,
                                                    (short)0, s1, false, false);
        s1 = __builtin_amdgcn_wmma_f32_16x16x32_f16(false, aq1, false, b1,
                                                    (short)0, s1, false, false);
      }

      // ---- online softmax: rows live in VGPR r (+8 in hi half) ----
      float corr[8];
#pragma unroll
      for (int r = 0; r < 8; ++r) {
        const float a0 = s0[r] * scale;
        const float a1 = s1[r] * scale;
        float mx = fmaxf(a0, a1);
#pragma unroll
        for (int m = 8; m >= 1; m >>= 1) mx = fmaxf(mx, __shfl_xor(mx, m, 16));
        const float nm = fmaxf(mrow[r], mx);
        corr[r] = __expf(mrow[r] - nm);
        mrow[r] = nm;
        const float p0 = __expf(a0 - nm);
        const float p1 = __expf(a1 - nm);
        float ps = p0 + p1;
#pragma unroll
        for (int m = 8; m >= 1; m >>= 1) ps += __shfl_xor(ps, m, 16);
        lsum[r] = lsum[r] * corr[r] + ps;
#pragma unroll
        for (int c = 0; c < 4; ++c) oo[c][r] *= corr[r];
        // P tile to LDS: row m = r + 8*hi, cols = this lane's key column.
        pb[wave][r + hi * 8][ln]      = (_Float16)p0;
        pb[wave][r + hi * 8][16 + ln] = (_Float16)p1;
      }
      // intra-wave LDS write->read ordering
      asm volatile("s_wait_dscnt 0" ::: "memory");

      // ---- O += P (16x32) * Vc_block (32x64): 4 WMMAs over d-chunks ----
      v16h ap;
#pragma unroll
      for (int e = 0; e < 16; ++e) ap[e] = pb[wave][ln][frag_k(e, hi)];
#pragma unroll
      for (int c = 0; c < 4; ++c) {
        v16h bv;
#pragma unroll
        for (int e = 0; e < 16; ++e)
          bv[e] = vs[frag_k(e, hi)][c * 16 + ln]; // K-dim = key row, N = d
        oo[c] = __builtin_amdgcn_wmma_f32_16x16x32_f16(false, ap, false, bv,
                                                       (short)0, oo[c],
                                                       false, false);
      }
    }
    __syncthreads(); // protect ks/vs before next block's staging
  }

  // ---- normalize and write context (B,H,40,64) ----
  if (wave < MT) {
#pragma unroll
    for (int c = 0; c < 4; ++c) {
#pragma unroll
      for (int r = 0; r < 8; ++r) {
        const int m = r + hi * 8;
        const int u = wave * 16 + m;
        if (u < U_) {
          out[(((size_t)bh * U_) + u) * D_ + c * 16 + ln] = oo[c][r] / lsum[r];
        }
      }
    }
  }
}

// ---------------- launch -------------------------------------------------
extern "C" void kernel_launch(void* const* d_in, const int* in_sizes, int n_in,
                              void* d_out, int out_size, void* d_ws,
                              size_t ws_size, hipStream_t stream) {
  const float* q  = (const float*)d_in[0];
  const float* k  = (const float*)d_in[1];
  const float* v  = (const float*)d_in[2];
  const int*  idx = (const int*)d_in[3];
  float* out = (float*)d_out;

  // ws layout: M (B*H*L f32) | top (B*H*U i32, padded to 8KB) | Vc (B*H*L*D f32)
  float* M   = (float*)d_ws;
  int*   top = (int*)((char*)d_ws + (size_t)B_ * H_ * L_ * 4);
  float* vc  = (float*)((char*)d_ws + (size_t)B_ * H_ * L_ * 4 + 8192);

  pa_measure_kernel<<<(B_ * H_ * L_) / 8, 256, 0, stream>>>(q, k, idx, M);
  pa_topk_kernel<<<B_ * H_, 256, 0, stream>>>(M, top);
  pa_cumsum_kernel<<<(B_ * H_ * D_ + 255) / 256, 256, 0, stream>>>(v, vc);
  pa_attn_kernel<<<B_ * H_, 128, 0, stream>>>(q, k, vc, top, out);
}
